// LowRankMixtureDensityNetwork_27324581937231
// MI455X (gfx1250) — matrix operand order repaired and checked by reading, not compiled, not run on hardware
//
#include <hip/hip_runtime.h>

typedef __attribute__((ext_vector_type(16))) __bf16 bf16x16;
typedef __attribute__((ext_vector_type(8)))  __bf16 bf16x8;
typedef __attribute__((ext_vector_type(8)))  float  f32x8;

// Padded projection width: 5136 -> 5376 = 21 * 256 (block N-tile)
#define NPAD 5376
#define NOUT 5136

// ---------------------------------------------------------------------------
// fp32 -> bf16 conversion (RNE via clang native cast)
// ---------------------------------------------------------------------------
__global__ void f32_to_bf16_k(const float* __restrict__ in, __bf16* __restrict__ out, int n) {
  int i = blockIdx.x * 256 + threadIdx.x;
  if (i < n) out[i] = (__bf16)in[i];
}

// Wout [5136,512] -> bf16 [NPAD,512] with zeroed padding rows
__global__ void cvt_pad_wout_k(const float* __restrict__ W, __bf16* __restrict__ out) {
  int i = blockIdx.x * 256 + threadIdx.x;       // over NPAD*512
  int n = i >> 9, k = i & 511;
  out[i] = (n < NOUT) ? (__bf16)W[(size_t)n * 512 + k] : (__bf16)0.0f;
}

// bout [5136] -> f32 [NPAD] zero-padded
__global__ void pad_bias_k(const float* __restrict__ b, float* __restrict__ out) {
  int i = blockIdx.x * 256 + threadIdx.x;
  if (i < NPAD) out[i] = (i < NOUT) ? b[i] : 0.f;
}

// ---------------------------------------------------------------------------
// WMMA fragment loads (guard-free; all dims padded to tile multiples).
// A operand 16x32 bf16 (ISA 7.12.2): lanes 0-15: row M=lane, K {0..7,16..23};
// lanes 16-31: row M=lane-16, K {8..15,24..31}. Vector elem i -> VGPR i/2.
// ---------------------------------------------------------------------------
__device__ inline bf16x16 load_a_frag(const __bf16* __restrict__ A, int lda,
                                      int m_base, int k0, int lane) {
  int row = m_base + (lane & 15);
  int kb  = (lane >> 4) * 8;
  const __bf16* p = A + (size_t)row * lda + k0 + kb;
  bf16x8 lo = *reinterpret_cast<const bf16x8*>(p);
  bf16x8 hi = *reinterpret_cast<const bf16x8*>(p + 16);
  bf16x16 r;
#pragma unroll
  for (int i = 0; i < 8; ++i) { r[i] = lo[i]; r[8 + i] = hi[i]; }
  return r;
}

// B operand 32x16 bf16: lanes 0-15: col N=lane, K {0..7,16..23}; lanes 16-31: col
// N=lane-16, K {8..15,24..31}. For C = X*W^T, column n of B == row n of W (K-contig).
__device__ inline bf16x16 load_b_frag(const __bf16* __restrict__ W, int ldw,
                                      int n_base, int k0, int lane) {
  int col = n_base + (lane & 15);
  int kb  = (lane >> 4) * 8;
  const __bf16* p = W + (size_t)col * ldw + k0 + kb;
  bf16x8 lo = *reinterpret_cast<const bf16x8*>(p);
  bf16x8 hi = *reinterpret_cast<const bf16x8*>(p + 16);
  bf16x16 r;
#pragma unroll
  for (int i = 0; i < 8; ++i) { r[i] = lo[i]; r[8 + i] = hi[i]; }
  return r;
}

__device__ inline float elu1(float x) { return x > 0.f ? x : (__expf(x) - 1.f); }

// ---------------------------------------------------------------------------
// Tiled bf16 GEMM: C[M,N] = act( A[M,K] * W[N,K]^T + bias ).
// Block = 128 threads = 4 waves; each wave owns a 64x64 tile (4x4 WMMA tiles,
// 16 WMMAs per 8 fragment loads per K-step). Grid: x = N/256, y = M/64.
// Requires M%64==0, N%256==0, K%32==0.
// ---------------------------------------------------------------------------
template <bool DO_ELU>
__global__ __launch_bounds__(128) void gemm_bf16_wmma(
    const __bf16* __restrict__ A, const __bf16* __restrict__ W,
    const float* __restrict__ bias, float* __restrict__ C,
    int M, int N, int K) {
  const int lane   = threadIdx.x & 31;
  const int wave   = threadIdx.x >> 5;
  const int m_base = blockIdx.y * 64;
  const int n_base = blockIdx.x * 256 + wave * 64;

  f32x8 acc[4][4];
  const f32x8 z = {};
#pragma unroll
  for (int i = 0; i < 4; ++i)
#pragma unroll
    for (int j = 0; j < 4; ++j) acc[i][j] = z;

  for (int k0 = 0; k0 < K; k0 += 32) {
    bf16x16 a[4], b[4];
#pragma unroll
    for (int i = 0; i < 4; ++i) a[i] = load_a_frag(A, K, m_base + 16 * i, k0, lane);
#pragma unroll
    for (int j = 0; j < 4; ++j) b[j] = load_b_frag(W, K, n_base + 16 * j, k0, lane);
#pragma unroll
    for (int i = 0; i < 4; ++i)
#pragma unroll
      for (int j = 0; j < 4; ++j)
        acc[i][j] = __builtin_amdgcn_wmma_f32_16x16x32_bf16(
            false, a[i], false, b[j], (short)0, acc[i][j], false, false);
  }

  // C/D layout: VGPR v, lane l -> row = v + 8*(l>>4), col = l&15.
  const int hi = lane >> 4;
  const int ln = lane & 15;
  float bv[4];
#pragma unroll
  for (int j = 0; j < 4; ++j) bv[j] = bias[n_base + 16 * j + ln];
#pragma unroll
  for (int i = 0; i < 4; ++i) {
    const int mr = m_base + 16 * i + hi * 8;
#pragma unroll
    for (int j = 0; j < 4; ++j) {
      const int nc = n_base + 16 * j + ln;
#pragma unroll
      for (int v = 0; v < 8; ++v) {
        float x = acc[i][j][v] + bv[j];
        if (DO_ELU) x = elu1(x);
        C[(size_t)(mr + v) * N + nc] = x;
      }
    }
  }
}

// ---------------------------------------------------------------------------
// BatchNorm stats, deterministic two-stage tree.
// Stage 1: grid (2, 32) x 256 threads; block (x,y) sums 256 rows for feature
// h = x*256+tid. Coalesced: consecutive lanes -> consecutive features.
// ---------------------------------------------------------------------------
__global__ void bn_stats1_k(const float* __restrict__ Y, float* __restrict__ ps,
                            float* __restrict__ ps2) {
  int h  = blockIdx.x * 256 + threadIdx.x;       // 0..511
  int r0 = blockIdx.y * 256;                     // 32 chunks of 256 rows
  float s = 0.f, s2 = 0.f;
  for (int b = 0; b < 256; ++b) {
    float x = Y[(size_t)(r0 + b) * 512 + h];
    s += x; s2 += x * x;
  }
  ps [(size_t)blockIdx.y * 512 + h] = s;
  ps2[(size_t)blockIdx.y * 512 + h] = s2;
}

// Stage 2: 512 threads combine the 32 chunks in fixed order.
__global__ void bn_stats2_k(const float* __restrict__ ps, const float* __restrict__ ps2,
                            float* __restrict__ mean, float* __restrict__ rstd) {
  int h = blockIdx.x * 256 + threadIdx.x;
  if (h >= 512) return;
  float s = 0.f, s2 = 0.f;
  for (int c = 0; c < 32; ++c) { s += ps[(size_t)c * 512 + h]; s2 += ps2[(size_t)c * 512 + h]; }
  float m = s * (1.f / 8192.f);
  float v = s2 * (1.f / 8192.f) - m * m;
  mean[h] = m;
  rstd[h] = rsqrtf(v + 1e-5f);
}

__global__ void bn_apply_k(const float* __restrict__ Y, const float* __restrict__ mean,
                           const float* __restrict__ rstd, const float* __restrict__ g,
                           const float* __restrict__ be, __bf16* __restrict__ X) {
  int i = blockIdx.x * 256 + threadIdx.x;  // over 8192*512
  int h = i & 511;
  float x = g[h] * (Y[i] - mean[h]) * rstd[h] + be[h];
  X[i] = (__bf16)x;
}

// ---------------------------------------------------------------------------
// MDN log-prob for one slice of 1024 rows. Block = 256 threads = 16 rows x 16
// components; thread (b,k) builds the 8x8 capacitance matrix, Cholesky, solve.
// 16-lane shuffle-logsumexp combines the K components.
// ---------------------------------------------------------------------------
__global__ __launch_bounds__(256) void mdn_logprob_k(
    const float* __restrict__ out, const float* __restrict__ data,
    float* __restrict__ logp, int row_base) {
  const int tid = threadIdx.x;
  const int bl  = tid >> 4;   // local row 0..15
  const int k   = tid & 15;   // component
  const int r   = row_base + blockIdx.x * 16 + bl;              // global row
  const float* o  = out + (size_t)(blockIdx.x * 16 + bl) * NPAD; // slice-local
  const float* dp = data + (size_t)r * 32;
  const float* mu = o + 16 + k * 32;
  const float* cv = o + 528 + k * 288;   // [0,32) log-diag, [32,288) factor [32][8]

  float cap[8][8], t[8];
#pragma unroll
  for (int i = 0; i < 8; ++i) {
    t[i] = 0.f;
#pragma unroll
    for (int j = 0; j <= i; ++j) cap[i][j] = 0.f;
  }
  float mah1 = 0.f, ldg = 0.f;
  for (int d = 0; d < 32; ++d) {
    float cd   = __expf(cv[d]) + 1e-5f;
    float sdi  = rsqrtf(cd);
    float diff = dp[d] - mu[d];
    float dsc  = diff * sdi;
    mah1 += dsc * dsc;
    ldg  += __logf(cd);
    const float* f = cv + 32 + d * 8;
    float a[8];
#pragma unroll
    for (int q = 0; q < 8; ++q) a[q] = f[q] * sdi;
#pragma unroll
    for (int i = 0; i < 8; ++i) {
      t[i] += a[i] * dsc;
#pragma unroll
      for (int j = 0; j <= i; ++j) cap[i][j] += a[i] * a[j];
    }
  }
#pragma unroll
  for (int i = 0; i < 8; ++i) cap[i][i] += 1.f;

  // In-place Cholesky of 8x8 (lower)
  float ld2 = 0.f;
#pragma unroll
  for (int j = 0; j < 8; ++j) {
    float s = cap[j][j];
#pragma unroll
    for (int p = 0; p < j; ++p) s -= cap[j][p] * cap[j][p];
    float dj = sqrtf(s);
    cap[j][j] = dj;
    ld2 += __logf(dj);
    float inv = 1.f / dj;
#pragma unroll
    for (int i = j + 1; i < 8; ++i) {
      float s2 = cap[i][j];
#pragma unroll
      for (int p = 0; p < j; ++p) s2 -= cap[i][p] * cap[j][p];
      cap[i][j] = s2 * inv;
    }
  }
  // Forward solve L y = t (overwrite t with y)
#pragma unroll
  for (int j = 0; j < 8; ++j) {
    float s = t[j];
#pragma unroll
    for (int p = 0; p < j; ++p) s -= cap[j][p] * t[p];
    t[j] = s / cap[j][j];
  }
  float yy = 0.f;
#pragma unroll
  for (int q = 0; q < 8; ++q) yy += t[q] * t[q];

  const float LOG2PI = 1.8378770664093453f;
  float logdet = ldg + 2.f * ld2;
  float mah    = mah1 - yy;
  float comp   = -0.5f * (32.f * LOG2PI + logdet + mah);
  float w      = o[k];

  // logsumexp over the 16 lanes sharing a row (width-16 xor shuffles)
  auto lse16 = [&](float v) -> float {
    float m = v;
#pragma unroll
    for (int off = 8; off > 0; off >>= 1) m = fmaxf(m, __shfl_xor(m, off, 16));
    float s = __expf(v - m);
#pragma unroll
    for (int off = 8; off > 0; off >>= 1) s += __shfl_xor(s, off, 16);
    return m + __logf(s);
  };
  // logsumexp_k(log_softmax(w) + comp) = lse(w+comp) - lse(w)
  float l1 = lse16(w + comp);
  float l0 = lse16(w);
  if (k == 0) logp[r] = l1 - l0;
}

// ---------------------------------------------------------------------------
// Final: -mean(logp), deterministic two-stage tree (no atomics).
// ---------------------------------------------------------------------------
__global__ void sum_stage1_k(const float* __restrict__ logp, float* __restrict__ part) {
  __shared__ float sm[256];
  int i = blockIdx.x * 256 + threadIdx.x;
  sm[threadIdx.x] = logp[i];
  __syncthreads();
  for (int s = 128; s > 0; s >>= 1) {
    if (threadIdx.x < s) sm[threadIdx.x] += sm[threadIdx.x + s];
    __syncthreads();
  }
  if (threadIdx.x == 0) part[blockIdx.x] = sm[0];
}

__global__ void sum_stage2_k(const float* __restrict__ part, float* __restrict__ outv) {
  float v = part[threadIdx.x];   // 32 lanes, one wave
#pragma unroll
  for (int off = 16; off > 0; off >>= 1) v += __shfl_xor(v, off, 32);
  if (threadIdx.x == 0) outv[0] = -v * (1.f / 8192.f);
}

// ---------------------------------------------------------------------------
// Host launcher
// ---------------------------------------------------------------------------
extern "C" void kernel_launch(void* const* d_in, const int* in_sizes, int n_in,
                              void* d_out, int out_size, void* d_ws, size_t ws_size,
                              hipStream_t stream) {
  (void)in_sizes; (void)n_in; (void)out_size; (void)ws_size;
  const float* data    = (const float*)d_in[0];   // [8192,32]
  const float* context = (const float*)d_in[1];   // [8192,128]
  const float* W0      = (const float*)d_in[2];   // [512,128]
  const float* b0      = (const float*)d_in[3];
  const float* g0      = (const float*)d_in[4];
  const float* be0     = (const float*)d_in[5];
  const float* Wh      = (const float*)d_in[6];   // [3,512,512]
  const float* bh      = (const float*)d_in[7];
  const float* gh      = (const float*)d_in[8];
  const float* beh     = (const float*)d_in[9];
  const float* Wout    = (const float*)d_in[10];  // [5136,512]
  const float* bout    = (const float*)d_in[11];
  float* outp = (float*)d_out;

  char* ws = (char*)d_ws;
  size_t off = 0;
  auto take = [&](size_t bytes) -> char* {
    char* p = ws + off;
    off = (off + bytes + 255) & ~(size_t)255;
    return p;
  };
  __bf16* cbf   = (__bf16*)take(8192ull * 128 * 2);
  __bf16* w0bf  = (__bf16*)take(512ull * 128 * 2);
  __bf16* whbf  = (__bf16*)take(3ull * 512 * 512 * 2);
  __bf16* wobf  = (__bf16*)take((size_t)NPAD * 512 * 2);
  float*  boutp = (float*)take((size_t)NPAD * 4);
  __bf16* xbf   = (__bf16*)take(8192ull * 512 * 2);
  float*  Y     = (float*)take(8192ull * 512 * 4);
  float*  mb    = (float*)take(512 * 4);
  float*  rb    = (float*)take(512 * 4);
  float*  ps    = (float*)take(32ull * 512 * 4);
  float*  ps2   = (float*)take(32ull * 512 * 4);
  float*  obuf  = (float*)take(1024ull * NPAD * 4);
  float*  logp  = (float*)take(8192 * 4);
  float*  part  = (float*)take(32 * 4);

  auto cvt = [&](const float* src, __bf16* dst, int n) {
    f32_to_bf16_k<<<(n + 255) / 256, 256, 0, stream>>>(src, dst, n);
  };
  cvt(context, cbf, 8192 * 128);
  cvt(W0,     w0bf, 512 * 128);
  cvt(Wh,     whbf, 3 * 512 * 512);
  cvt_pad_wout_k<<<(NPAD * 512) / 256, 256, 0, stream>>>(Wout, wobf);
  pad_bias_k<<<(NPAD + 255) / 256, 256, 0, stream>>>(bout, boutp);

  // Layer 0: [8192,128] x [512,128]^T, ELU, BN
  gemm_bf16_wmma<true><<<dim3(2, 128), 128, 0, stream>>>(cbf, w0bf, b0, Y, 8192, 512, 128);
  bn_stats1_k<<<dim3(2, 32), 256, 0, stream>>>(Y, ps, ps2);
  bn_stats2_k<<<2, 256, 0, stream>>>(ps, ps2, mb, rb);
  bn_apply_k<<<(8192 * 512) / 256, 256, 0, stream>>>(Y, mb, rb, g0, be0, xbf);

  // Hidden layers
  for (int i = 0; i < 3; ++i) {
    gemm_bf16_wmma<true><<<dim3(2, 128), 128, 0, stream>>>(
        xbf, whbf + (size_t)i * 512 * 512, bh + i * 512, Y, 8192, 512, 512);
    bn_stats1_k<<<dim3(2, 32), 256, 0, stream>>>(Y, ps, ps2);
    bn_stats2_k<<<2, 256, 0, stream>>>(ps, ps2, mb, rb);
    bn_apply_k<<<(8192 * 512) / 256, 256, 0, stream>>>(Y, mb, rb, gh + i * 512, beh + i * 512, xbf);
  }

  // Output projection (padded to NPAD cols) + MDN, 8 slices of 1024 rows
  for (int s = 0; s < 8; ++s) {
    gemm_bf16_wmma<false><<<dim3(NPAD / 256, 16), 128, 0, stream>>>(
        xbf + (size_t)s * 1024 * 512, wobf, boutp, obuf, 1024, NPAD, 512);
    mdn_logprob_k<<<64, 256, 0, stream>>>(obuf, data, logp, s * 1024);
  }

  sum_stage1_k<<<32, 256, 0, stream>>>(logp, part);
  sum_stage2_k<<<1, 32, 0, stream>>>(part, outp);
}
